// GaussianRasterizer_50397146251309
// MI455X (gfx1250) — matrix operand order repaired and checked by reading, not compiled, not run on hardware
//
#include <hip/hip_runtime.h>
#include <hip/hip_bf16.h>

typedef __attribute__((ext_vector_type(16))) _Float16 v16h;
typedef __attribute__((ext_vector_type(8)))  _Float16 v8h;
typedef __attribute__((ext_vector_type(8)))  float    v8f;

#define NP 512
#define IMG_W 256
#define IMG_H 256
#define NPIX (IMG_W * IMG_H)

// d_out layout (floats): color[3*NPIX] | radii[NP] | depth[NPIX] | opac[NPIX] | n_touched[NP]
#define OUT_COLOR  0
#define OUT_RADII  (3 * NPIX)
#define OUT_DEPTH  (OUT_RADII + NP)
#define OUT_OPAC   (OUT_DEPTH + NPIX)
#define OUT_NTOUCH (OUT_OPAC + NPIX)

// ws layout (float offsets)
#define WS_DEP   0
#define WS_UPX   (1 * NP)
#define WS_UPY   (2 * NP)
#define WS_UCA   (3 * NP)
#define WS_UCB   (4 * NP)
#define WS_UCC   (5 * NP)
#define WS_UOP   (6 * NP)
#define WS_ORDER (7 * NP)          // int[NP]
#define WS_SP4   (8 * NP)          // float4[NP]: px,py,ca,cb
#define WS_SP2   (12 * NP)         // float2[NP]: cc,op
#define WS_STAGE (14 * NP)         // _Float16[4][NP]  (channel-major) == 2*NP floats

// ---------------------------------------------------------------------------
// Kernel 1: per-gaussian preprocess (conic, radii, projected center, validity)
// ---------------------------------------------------------------------------
__global__ __launch_bounds__(256) void gs_preprocess(
    const float* __restrict__ means, const float* __restrict__ scales,
    const float* __restrict__ rots,  const float* __restrict__ opac,
    const float* __restrict__ vm,    const float* __restrict__ pm,
    float* __restrict__ ws, float* __restrict__ out)
{
    int i = blockIdx.x * blockDim.x + threadIdx.x;
    if (i >= NP) return;

    float m0 = means[i*3+0], m1 = means[i*3+1], m2 = means[i*3+2];

    // p_view = Rw*m + t  (Rw = vm[:3,:3] row-major, t = vm[:3,3])
    float pv0 = vm[0]*m0 + vm[1]*m1 + vm[2]*m2  + vm[3];
    float pv1 = vm[4]*m0 + vm[5]*m1 + vm[6]*m2  + vm[7];
    float pv2 = vm[8]*m0 + vm[9]*m1 + vm[10]*m2 + vm[11];
    float depth = pv2;

    // p_hom = projmatrix @ [m,1]
    float ph0 = pm[0]*m0  + pm[1]*m1  + pm[2]*m2  + pm[3];
    float ph1 = pm[4]*m0  + pm[5]*m1  + pm[6]*m2  + pm[7];
    float ph3 = pm[12]*m0 + pm[13]*m1 + pm[14]*m2 + pm[15];
    float pw  = 1.0f / (ph3 + 1e-7f);
    float pp0 = ph0 * pw, pp1 = ph1 * pw;

    // quaternion -> rotation
    float q0 = rots[i*4+0], q1 = rots[i*4+1], q2 = rots[i*4+2], q3 = rots[i*4+3];
    float qn = rsqrtf(q0*q0 + q1*q1 + q2*q2 + q3*q3);
    float r = q0*qn, x = q1*qn, y = q2*qn, z = q3*qn;
    float R00 = 1.f - 2.f*(y*y + z*z), R01 = 2.f*(x*y - r*z), R02 = 2.f*(x*z + r*y);
    float R10 = 2.f*(x*y + r*z), R11 = 1.f - 2.f*(x*x + z*z), R12 = 2.f*(y*z - r*x);
    float R20 = 2.f*(x*z - r*y), R21 = 2.f*(y*z + r*x), R22 = 1.f - 2.f*(x*x + y*y);

    float s0 = scales[i*3+0], s1 = scales[i*3+1], s2 = scales[i*3+2];
    float s0q = s0*s0, s1q = s1*s1, s2q = s2*s2;

    // cov3d = (R*diag(s)) (R*diag(s))^T
    float C00 = R00*R00*s0q + R01*R01*s1q + R02*R02*s2q;
    float C01 = R00*R10*s0q + R01*R11*s1q + R02*R12*s2q;
    float C02 = R00*R20*s0q + R01*R21*s1q + R02*R22*s2q;
    float C11 = R10*R10*s0q + R11*R11*s1q + R12*R12*s2q;
    float C12 = R10*R20*s0q + R11*R21*s1q + R12*R22*s2q;
    float C22 = R20*R20*s0q + R21*R21*s1q + R22*R22*s2q;

    // Jacobian
    float tz = (fabsf(depth) < 1e-6f) ? 1e-6f : depth;
    const float lim = 1.3f * 0.5f;                 // 1.3 * tanfov
    float tx = fminf(fmaxf(pv0 / tz, -lim), lim) * tz;
    float ty = fminf(fmaxf(pv1 / tz, -lim), lim) * tz;
    const float FXc = 256.0f, FYc = 256.0f;        // W/(2*tanfovx), H/(2*tanfovy)
    float itz = 1.0f / tz;
    float J00 = FXc * itz, J02 = -FXc * tx * itz * itz;
    float J11 = FYc * itz, J12 = -FYc * ty * itz * itz;

    // T2 = J @ Rw (2x3)
    float T00 = J00*vm[0] + J02*vm[8];
    float T01 = J00*vm[1] + J02*vm[9];
    float T02 = J00*vm[2] + J02*vm[10];
    float T10 = J11*vm[4] + J12*vm[8];
    float T11 = J11*vm[5] + J12*vm[9];
    float T12 = J11*vm[6] + J12*vm[10];

    // cov2d = T2 C T2^T
    float A0 = T00*C00 + T01*C01 + T02*C02;
    float A1 = T00*C01 + T01*C11 + T02*C12;
    float A2 = T00*C02 + T01*C12 + T02*C22;
    float B0 = T10*C00 + T11*C01 + T12*C02;
    float B1 = T10*C01 + T11*C11 + T12*C12;
    float B2 = T10*C02 + T11*C12 + T12*C22;
    float c00 = A0*T00 + A1*T01 + A2*T02 + 0.3f;
    float c01 = A0*T10 + A1*T11 + A2*T12;
    float c11 = B0*T10 + B1*T11 + B2*T12 + 0.3f;

    float det = c00*c11 - c01*c01;
    bool valid = (depth > 0.2f) && (det > 0.0f);
    float det_s = valid ? det : 1.0f;
    float ca =  c11 / det_s;
    float cb = -c01 / det_s;
    float cc =  c00 / det_s;
    float mid = 0.5f * (c00 + c11);
    float lam = mid + sqrtf(fmaxf(mid*mid - det, 0.1f));
    float radf = valid ? ceilf(3.0f * sqrtf(lam)) : 0.0f;

    float px = ((pp0 + 1.0f) * (float)IMG_W - 1.0f) * 0.5f;
    float py = ((pp1 + 1.0f) * (float)IMG_H - 1.0f) * 0.5f;

    ws[WS_DEP + i] = depth;
    ws[WS_UPX + i] = px;
    ws[WS_UPY + i] = py;
    ws[WS_UCA + i] = ca;
    ws[WS_UCB + i] = cb;
    ws[WS_UCC + i] = cc;
    ws[WS_UOP + i] = valid ? opac[i] : 0.0f;   // invalid => alpha 0 => excluded

    out[OUT_RADII  + i] = radf;
    out[OUT_NTOUCH + i] = 0.0f;                // zero accumulator
}

// ---------------------------------------------------------------------------
// Kernel 2: bitonic depth-sort of 512 gaussians + gather into packed arrays
// ---------------------------------------------------------------------------
__global__ __launch_bounds__(512) void gs_sort_gather(
    const float* __restrict__ colors, float* __restrict__ ws)
{
    __shared__ float sd[NP];
    __shared__ int   si[NP];
    int tid = threadIdx.x;
    sd[tid] = ws[WS_DEP + tid];
    si[tid] = tid;
    __syncthreads();

    for (int k = 2; k <= NP; k <<= 1) {
        for (int j = k >> 1; j > 0; j >>= 1) {
            int ixj = tid ^ j;
            if (ixj > tid) {
                bool up = ((tid & k) == 0);
                float d0 = sd[tid], d1 = sd[ixj];
                if ((d0 > d1) == up) {
                    sd[tid] = d1; sd[ixj] = d0;
                    int t = si[tid]; si[tid] = si[ixj]; si[ixj] = t;
                }
            }
            __syncthreads();
        }
    }

    int g = si[tid];
    ((int*)(ws + WS_ORDER))[tid] = g;
    float4* p4 = (float4*)(ws + WS_SP4);
    float2* p2 = (float2*)(ws + WS_SP2);
    p4[tid] = make_float4(ws[WS_UPX + g], ws[WS_UPY + g],
                          ws[WS_UCA + g], ws[WS_UCB + g]);
    p2[tid] = make_float2(ws[WS_UCC + g], ws[WS_UOP + g]);
    // channel-major staging: stage[ch][g] so A-fragments are contiguous in K
    _Float16* stage = (_Float16*)(ws + WS_STAGE);
    stage[0 * NP + tid] = (_Float16)colors[g*3 + 0];
    stage[1 * NP + tid] = (_Float16)colors[g*3 + 1];
    stage[2 * NP + tid] = (_Float16)colors[g*3 + 2];
    stage[3 * NP + tid] = (_Float16)sd[tid];       // sorted depth, channel 3
}

// ---------------------------------------------------------------------------
// Kernel 3: tiled WMMA rasterizer.
//  block = 256 threads = 8 waves; wave handles 16 pixels; 512 blocks.
//  Per chunk of 32 gaussians: D(16ch x 16pix) += A(ch x K) * B(K x pix)
//  via v_wmma_f32_16x16x32_f16, fully unrolled (16 WMMAs / wave).
//  n_touched: 16 ballots -> uniform SGPR masks, cndmask select chain routes
//  mask j to lanes j / j+16, then ONE lane-distinct ds_add_u32 per chunk.
// ---------------------------------------------------------------------------
__global__ __launch_bounds__(256) void gs_rasterize(
    const float* __restrict__ ws, const float* __restrict__ bg,
    float* __restrict__ out)
{
    __shared__ __align__(16) float4 sp4[NP];            // px,py,ca,cb
    __shared__ __align__(8)  float2 sp2[NP];            // cc,op
    __shared__ __align__(16) _Float16 sstage[4 * NP];   // [ch][g]
    __shared__ int scnt[NP];

    const int tid = threadIdx.x;
    const float4* g_p4 = (const float4*)(ws + WS_SP4);
    const float2* g_p2 = (const float2*)(ws + WS_SP2);
    const _Float16* g_stage = (const _Float16*)(ws + WS_STAGE);

    __builtin_prefetch(g_p4 + tid, 0, 0);    // global_prefetch_b8 into caches

    for (int i = tid; i < NP; i += 256) {
        sp4[i] = g_p4[i];
        sp2[i] = g_p2[i];
        scnt[i] = 0;
    }
    for (int i = tid; i < 4 * NP; i += 256) sstage[i] = g_stage[i];
    __syncthreads();

    const int wave = tid >> 5;
    const int lane = tid & 31;
    const bool hi  = (lane & 16) != 0;
    const int ch   = lane & 15;                       // A-matrix row (channel)
    const int mypix = blockIdx.x * 128 + wave * 16 + (lane & 15);
    const float fx = (float)(mypix & (IMG_W - 1));
    const float fy = (float)(mypix >> 8);

    float T = 1.0f;                                   // running transmittance
    v8f acc = {};

#pragma unroll
    for (int chunk = 0; chunk < 16; ++chunk) {
        const int k0 = chunk * 32;
        const int kb = k0 + (hi ? 16 : 0);            // my 16-long K run

        float a_arr[16];
        unsigned masks[16];
#pragma unroll
        for (int j = 0; j < 16; ++j) {
            int g = kb + j;
            float4 q4 = sp4[g];                       // px,py,ca,cb (b128)
            float2 q2 = sp2[g];                       // cc,op       (b64)
            float dx = q4.x - fx;
            float dy = q4.y - fy;
            float power = -0.5f * (q4.z*dx*dx + q2.x*dy*dy) - q4.w*dx*dy;
            float al = fminf(0.99f, q2.y * __expf(power));
            bool keep = (power <= 0.0f) && (al >= (1.0f / 255.0f));
            masks[j] = (unsigned)__ballot(keep);      // wave-uniform SGPR
            a_arr[j] = keep ? al : 0.0f;
        }

        // in-lane prefix over my K run
        float wtmp[16];
        float t = 1.0f;
#pragma unroll
        for (int j = 0; j < 16; ++j) {
            wtmp[j] = t * a_arr[j];
            t *= (1.0f - a_arr[j]);
        }
        float t_other = __shfl_xor(t, 16, 32);        // other half's full product
        float pre = T * (hi ? t_other : 1.0f);        // hi lanes see lo's product

        v16h wv;                                      // B matrix: K x pixels
#pragma unroll
        for (int j = 0; j < 16; ++j) wv[j] = (_Float16)(wtmp[j] * pre);
        T *= t * t_other;                             // advance transmittance

        // A matrix: rows = channels (r,g,b,depth,0...), contiguous K (k = 16h + s), as in B.
        // Channel-major staging => two ds_load_b128 + concat.
        v16h av = {};
        if (ch < 4) {
            const int baseA = hi ? (k0 + 16) : k0;    // K 0-15 / 16-31 of chunk
            v8h lo8 = *(const v8h*)&sstage[ch * NP + baseA];        // K run
            v8h hi8 = *(const v8h*)&sstage[ch * NP + baseA + 8];    // K run +8
            av = __builtin_shufflevector(lo8, hi8,
                                         0, 1, 2, 3, 4, 5, 6, 7,
                                         8, 9, 10, 11, 12, 13, 14, 15);
        }

        acc = __builtin_amdgcn_wmma_f32_16x16x32_f16(
                  false, av, false, wv, (short)0, acc, false, false);

        // n_touched: select chain routes uniform mask[ch] to each lane;
        // lane L<16 counts gaussian k0+L (low half-mask), lane L>=16 counts
        // gaussian k0+L (high half-mask). Lane-distinct addresses => plain
        // ds_add_u32, no branches, no atomic-optimizer expansion.
        unsigned msel = masks[0];
#pragma unroll
        for (int j = 1; j < 16; ++j) msel = (ch == j) ? masks[j] : msel;
        unsigned half_mask = hi ? (msel >> 16) : (msel & 0xFFFFu);
        atomicAdd(&scnt[k0 + lane], (int)__popc(half_mask));
    }

    // lane L<16 holds channels 0..7 for pixel (base+L) in acc[0..7]
    if (lane < 16) {
        float b0 = bg[0], b1 = bg[1], b2 = bg[2];
        out[OUT_COLOR + 0 * NPIX + mypix] = acc[0] + T * b0;
        out[OUT_COLOR + 1 * NPIX + mypix] = acc[1] + T * b1;
        out[OUT_COLOR + 2 * NPIX + mypix] = acc[2] + T * b2;
        out[OUT_DEPTH + mypix] = acc[3];
        out[OUT_OPAC  + mypix] = 1.0f - T;
    }

    __syncthreads();
    const int* order = (const int*)(ws + WS_ORDER);
    for (int i = tid; i < NP; i += 256) {
        int c = scnt[i];
        if (c) atomicAdd(&out[OUT_NTOUCH + order[i]], (float)c);
    }
}

// ---------------------------------------------------------------------------
extern "C" void kernel_launch(void* const* d_in, const int* in_sizes, int n_in,
                              void* d_out, int out_size, void* d_ws, size_t ws_size,
                              hipStream_t stream)
{
    const float* means  = (const float*)d_in[0];
    const float* scales = (const float*)d_in[1];
    const float* rots   = (const float*)d_in[2];
    const float* opac   = (const float*)d_in[3];
    const float* colors = (const float*)d_in[4];
    const float* vm     = (const float*)d_in[5];
    const float* pm     = (const float*)d_in[6];
    // d_in[7] = campos (unused by reference math), d_in[8] = bg
    const float* bg     = (const float*)d_in[8];

    float* out = (float*)d_out;
    float* ws  = (float*)d_ws;

    gs_preprocess<<<dim3(2), dim3(256), 0, stream>>>(means, scales, rots, opac,
                                                     vm, pm, ws, out);
    gs_sort_gather<<<dim3(1), dim3(512), 0, stream>>>(colors, ws);
    gs_rasterize<<<dim3(512), dim3(256), 0, stream>>>(ws, bg, out);
}
